// DistributedLinear_16827681866139
// MI455X (gfx1250) — compile-verified, matrix operand order checked by reading, not run on hardware
//
#include <hip/hip_runtime.h>

typedef __attribute__((ext_vector_type(16))) _Float16 v16h;
typedef __attribute__((ext_vector_type(8)))  _Float16 v8h;
typedef __attribute__((ext_vector_type(8)))  float    v8f;
typedef int v4i __attribute__((vector_size(16)));   // matches builtin param type

#define M_DIM 8192
#define K_DIM 4096
#define N_DIM 16384

constexpr int BM = 128;         // block tile M
constexpr int BN = 256;         // block tile N
constexpr int BK = 32;          // K-step (one WMMA K per step)
constexpr int SK = BK + 8;      // padded LDS stride in halves (80B, 16B-aligned)
constexpr int KT = K_DIM / BK;  // 128 K iterations

// ---------------------------------------------------------------------------
// f32 -> f16 bulk conversion (8 elements / thread, 128-bit loads & stores)
// ---------------------------------------------------------------------------
__global__ __launch_bounds__(256) void cvt_f32_f16(const float* __restrict__ in,
                                                   _Float16* __restrict__ out,
                                                   long n) {
  long i = ((long)blockIdx.x * blockDim.x + threadIdx.x) * 8;
  if (i + 8 > n) return;
  float4 f0 = *(const float4*)(in + i);
  float4 f1 = *(const float4*)(in + i + 4);
  v8h o;
  o[0] = (_Float16)f0.x; o[1] = (_Float16)f0.y;
  o[2] = (_Float16)f0.z; o[3] = (_Float16)f0.w;
  o[4] = (_Float16)f1.x; o[5] = (_Float16)f1.y;
  o[6] = (_Float16)f1.z; o[7] = (_Float16)f1.w;
  *(v8h*)(out + i) = o;
}

// ---------------------------------------------------------------------------
// CDNA5 async memory->LDS copy (16 bytes / lane), ASYNCcnt-tracked.
// ---------------------------------------------------------------------------
__device__ __forceinline__ void async_cp16(const _Float16* g, _Float16* l) {
#if __has_builtin(__builtin_amdgcn_global_load_async_to_lds_b128)
  __builtin_amdgcn_global_load_async_to_lds_b128(
      (__attribute__((address_space(1))) v4i*)g,
      (__attribute__((address_space(3))) v4i*)l, 0, 0);
#else
  *(v8h*)l = *(const v8h*)g;  // fallback: VGPR round-trip
#endif
}

__device__ __forceinline__ void wait_async0() {
#if __has_builtin(__builtin_amdgcn_s_wait_asynccnt)
  __builtin_amdgcn_s_wait_asynccnt(0);
#elif __has_builtin(__builtin_amdgcn_global_load_async_to_lds_b128)
  asm volatile("s_wait_asynccnt 0x0" ::: "memory");
#endif
}

// Fragment assembled from two 16-byte LDS loads
union AB {
  v16h v;
  struct { v8h lo; v8h hi; } p;
};

// ---------------------------------------------------------------------------
// NT GEMM: C[m][n] = sum_k A[m][k]*B[n][k] + bias[n]
// A: [M][K] f16, B: [N][K] f16, C: [M][N] f32
// Block tile 128x256, 8 waves (2x4), wave tile 64x64 (16 WMMAs / K-slice).
// Double buffering via pointer swap; K-loop kept at unroll 1 so only one
// fragment generation is live (keeps VGPR count < 256, no MSB shuffling).
// ---------------------------------------------------------------------------
__global__ __launch_bounds__(256) void gemm_f16_wmma(
    const _Float16* __restrict__ A,
    const _Float16* __restrict__ B,
    const float* __restrict__ bias,
    float* __restrict__ C) {
  __shared__ __align__(16) _Float16 As[2][BM * SK];  // 2 x 10 KiB
  __shared__ __align__(16) _Float16 Bs[2][BN * SK];  // 2 x 20 KiB

  const int tid  = threadIdx.x;
  const int lane = tid & 31;
  const int wave = tid >> 5;     // 0..7
  const int wm   = wave >> 2;    // 0..1 : 64-row slab
  const int wn   = wave & 3;     // 0..3 : 64-col slab

  const int blockM = blockIdx.y * BM;
  const int blockN = blockIdx.x * BN;

  // Staging assignment: 256 threads; rows split into 64-row groups,
  // 4 chunks of 8 halves per 32-half row slice.
  const int srow = tid >> 2;        // 0..63
  const int scol = (tid & 3) * 8;   // 0,8,16,24

  const _Float16* gA0 = A + (size_t)(blockM + srow) * K_DIM + scol;  // +64*K for 2nd
  const _Float16* gB0 = B + (size_t)(blockN + srow) * K_DIM + scol;  // +q*64*K

  const int lStage = srow * SK + scol;   // per-thread LDS staging offset

  // Double-buffer pointers (swapped each iteration)
  _Float16* cA = &As[0][0];
  _Float16* nA = &As[1][0];
  _Float16* cB = &Bs[0][0];
  _Float16* nB = &Bs[1][0];

  // Prologue: stage K-slice 0 into current buffers
  async_cp16(gA0,                      cA + lStage);
  async_cp16(gA0 + (size_t)64 * K_DIM, cA + lStage + 64 * SK);
#pragma unroll
  for (int q = 0; q < 4; ++q)
    async_cp16(gB0 + (size_t)q * 64 * K_DIM, cB + lStage + q * 64 * SK);
  wait_async0();
  __syncthreads();

  v8f acc[4][4] = {};

  // WMMA 16-bit A/B operand addressing (identical for A and B in NT form):
  // lane L holds row (L&15); halves 0..7 = K(base + (L>>4)*8),
  // halves 8..15 = same + 16.
  const int lrow  = lane & 15;
  const int lks   = (lane >> 4) * 8;             // half offset 0 or 8
  const int fAoff = (wm * 64 + lrow) * SK + lks; // + i*16*SK per m-tile
  const int fBoff = (wn * 64 + lrow) * SK + lks; // + j*16*SK per n-tile

  auto compute = [&](const _Float16* pA, const _Float16* pB) {
    AB afrag[4];
#pragma unroll
    for (int i = 0; i < 4; ++i) {
      const _Float16* src = pA + fAoff + i * 16 * SK;
      afrag[i].p.lo = *(const v8h*)(src);
      afrag[i].p.hi = *(const v8h*)(src + 16);
    }
#pragma unroll
    for (int j = 0; j < 4; ++j) {
      AB bfrag;
      const _Float16* src = pB + fBoff + j * 16 * SK;
      bfrag.p.lo = *(const v8h*)(src);
      bfrag.p.hi = *(const v8h*)(src + 16);
#pragma unroll
      for (int i = 0; i < 4; ++i)
        acc[i][j] = __builtin_amdgcn_wmma_f32_16x16x32_f16(
            /*neg_a=*/false, afrag[i].v, /*neg_b=*/false, bfrag.v,
            /*c_mod=*/(short)0, acc[i][j], /*reuse_a=*/false, /*reuse_b=*/false);
    }
  };

#pragma unroll 1
  for (int kk = 0; kk < KT - 1; ++kk) {
    // Stage next K-slice into the other buffer (its readers passed the
    // previous barrier).
    const size_t off = (size_t)(kk + 1) * BK;
    async_cp16(gA0 + off,                      nA + lStage);
    async_cp16(gA0 + off + (size_t)64 * K_DIM, nA + lStage + 64 * SK);
#pragma unroll
    for (int q = 0; q < 4; ++q)
      async_cp16(gB0 + off + (size_t)q * 64 * K_DIM, nB + lStage + q * 64 * SK);

    compute(cA, cB);

    wait_async0();    // our async writes to next buffer landed in LDS
    __syncthreads();  // everyone's writes visible / reads done

    _Float16* t;
    t = cA; cA = nA; nA = t;
    t = cB; cB = nB; nB = t;
  }
  compute(cA, cB);  // last K-slice (nothing left to stage)

  // Epilogue: C/D layout -> lane holds N = lane&15, M = (lane>>4)*8 + vgpr
  const int nc = lane & 15;
  const int mr = (lane >> 4) * 8;
#pragma unroll
  for (int j = 0; j < 4; ++j) {
    const int n = blockN + wn * 64 + j * 16 + nc;
    const float bv = bias[n];
#pragma unroll
    for (int i = 0; i < 4; ++i) {
      const int m0 = blockM + wm * 64 + i * 16 + mr;
#pragma unroll
      for (int e = 0; e < 8; ++e) {
        C[(size_t)(m0 + e) * N_DIM + n] = acc[i][j][e] + bv;
      }
    }
  }
}

// ---------------------------------------------------------------------------
extern "C" void kernel_launch(void* const* d_in, const int* in_sizes, int n_in,
                              void* d_out, int out_size, void* d_ws, size_t ws_size,
                              hipStream_t stream) {
  (void)in_sizes; (void)n_in; (void)out_size; (void)ws_size;

  const float* x = (const float*)d_in[0];   // [M][K] f32
  const float* w = (const float*)d_in[1];   // [N][K] f32
  const float* b = (const float*)d_in[2];   // [N]    f32
  float* out = (float*)d_out;               // [M][N] f32

  // Workspace: fp16 copies of x and w (64 MiB + 128 MiB)
  _Float16* xh = (_Float16*)d_ws;
  _Float16* wh = xh + (size_t)M_DIM * K_DIM;

  const long nx = (long)M_DIM * K_DIM;      // 33,554,432
  const long nw = (long)N_DIM * K_DIM;      // 67,108,864

  cvt_f32_f16<<<(int)(nx / (8 * 256)), 256, 0, stream>>>(x, xh, nx);
  cvt_f32_f16<<<(int)(nw / (8 * 256)), 256, 0, stream>>>(w, wh, nw);

  dim3 grid(N_DIM / BN, M_DIM / BM);        // 64 x 64 blocks
  gemm_f16_wmma<<<grid, 256, 0, stream>>>(xh, wh, b, out);
}
